// WaveNet_51058571215001
// MI455X (gfx1250) — matrix operand order repaired and checked by reading, not compiled, not run on hardware
//
#include <hip/hip_runtime.h>
#include <hip/hip_bf16.h>
#include <math.h>
#include <stdint.h>

typedef __attribute__((ext_vector_type(4)))  __bf16 v4bf;
typedef __attribute__((ext_vector_type(8)))  __bf16 v8bf;
typedef __attribute__((ext_vector_type(16))) __bf16 v16bf;
typedef __attribute__((ext_vector_type(8)))  float  v8f;

#define SQRT_HALF 0.70710678118654752f

__device__ __forceinline__ v16bf cat16(v8bf a, v8bf b) {
  return __builtin_shufflevector(a, b, 0,1,2,3,4,5,6,7,8,9,10,11,12,13,14,15);
}

// flat->LDS offset: LDS aperture keeps the group-segment offset in addr[31:0]
__device__ __forceinline__ uint32_t lds_off(const void* p) {
  return (uint32_t)(uintptr_t)p;
}

// ---- dynamic LDS layout (bytes) ----
// xbuf : 80 rows * 136 bf16  = 21760   (x rows t0-16..t0+63, padded stride)
// lgbuf: 64 rows * 168 bf16  = 21504   (lc|gc|zero-pad; aliased by glds after GEMM1)
// bbuf : 2 * 256 rows * 40 bf16 = 40960 (double-buffered weight tiles, 80B stride)
#define XBUF_OFF   0
#define LGBUF_OFF  21760
#define BBUF_OFF   (21760 + 21504)
#define BBUF_BYTES 20480
#define SMEM_BYTES (BBUF_OFF + 2 * BBUF_BYTES)   // 84224

// Stage 256 rows x 32 k of bf16 weights (row stride kst elements in global)
// into LDS buffer at lds_base with padded row stride 40 elements, via async DMA.
__device__ __forceinline__ void stage_async(const __bf16* __restrict__ wsrc,
                                            int kst, int k0,
                                            uint32_t lds_base, int tid) {
  const __bf16* g = wsrc + (size_t)tid * kst + k0;   // this thread's row
  uint32_t la = lds_base + (uint32_t)tid * 80;
  #pragma unroll
  for (int j = 0; j < 4; ++j) {
    uint64_t ga = (uint64_t)(uintptr_t)(g + j * 8);
    uint32_t ld = la + (uint32_t)j * 16;
    asm volatile("global_load_async_to_lds_b128 %0, %1, off"
                 :: "v"(ld), "v"(ga) : "memory");
  }
}

__device__ __forceinline__ void wait_async_barrier() {
  asm volatile("s_wait_asynccnt 0x0" ::: "memory");
  __syncthreads();
}

// ---------------- weight-normalization prep: build bf16 [N][K] weights ----------------
// w1: [256][544]  cols 0..383 conv (tap-major), 384..463 lc, 464..527 gc, 528..543 zero
// w2: [256][128]  rows 0..127 skip weights (transposed), 128..255 out weights
__global__ __launch_bounds__(128) void wn_prep(
    const float* __restrict__ conv_k, const float* __restrict__ conv_g,
    const float* __restrict__ lc_k,   const float* __restrict__ lc_g,
    const float* __restrict__ gc_k,   const float* __restrict__ gc_g,
    const float* __restrict__ skip_k, const float* __restrict__ skip_g,
    const float* __restrict__ out_k,  const float* __restrict__ out_g,
    __bf16* __restrict__ w1, __bf16* __restrict__ w2)
{
  const int n   = blockIdx.x;
  const int tid = threadIdx.x;
  __shared__ float red[128];

  auto reduce = [&](float v) -> float {
    red[tid] = v; __syncthreads();
    for (int s = 64; s > 0; s >>= 1) {
      if (tid < s) red[tid] += red[tid + s];
      __syncthreads();
    }
    float r = red[0]; __syncthreads();
    return r;
  };

  float ss = 0.f;
  for (int i = tid; i < 384; i += 128) { float v = conv_k[i*256 + n]; ss += v*v; }
  float sc = conv_g[n] * rsqrtf(fmaxf(reduce(ss), 1e-12f));
  for (int i = tid; i < 384; i += 128)
    w1[(size_t)n*544 + i] = (__bf16)(conv_k[i*256 + n] * sc);

  ss = (tid < 80) ? lc_k[tid*256 + n] * lc_k[tid*256 + n] : 0.f;
  sc = lc_g[n] * rsqrtf(fmaxf(reduce(ss), 1e-12f));
  if (tid < 80) w1[(size_t)n*544 + 384 + tid] = (__bf16)(lc_k[tid*256 + n] * sc);

  ss = (tid < 64) ? gc_k[tid*256 + n] * gc_k[tid*256 + n] : 0.f;
  sc = gc_g[n] * rsqrtf(fmaxf(reduce(ss), 1e-12f));
  if (tid < 64) w1[(size_t)n*544 + 464 + tid] = (__bf16)(gc_k[tid*256 + n] * sc);

  if (tid < 16) w1[(size_t)n*544 + 528 + tid] = (__bf16)0.f;

  const float* src = (n < 128) ? skip_k : out_k;
  const int    col = (n < 128) ? n : (n - 128);
  const float  g   = (n < 128) ? skip_g[n] : out_g[n - 128];
  float v = src[tid*128 + col];
  sc = g * rsqrtf(fmaxf(reduce(v*v), 1e-12f));
  w2[(size_t)n*128 + tid] = (__bf16)(src[tid*128 + col] * sc);
}

// ---------------- fused WaveNet residual block ----------------
__global__ __launch_bounds__(256) void wavenet_main(
    const float* __restrict__ x, const float* __restrict__ lcond,
    const float* __restrict__ gcond,
    const float* __restrict__ conv_bias, const float* __restrict__ skip_bias,
    const float* __restrict__ out_bias,
    const __bf16* __restrict__ w1, const __bf16* __restrict__ w2,
    float* __restrict__ dout)
{
  constexpr int       T   = 16384;
  constexpr long long TOT = 8LL * 16384 * 128;

  extern __shared__ char smem[];
  __bf16* xbuf  = (__bf16*)(smem + XBUF_OFF);
  __bf16* lgbuf = (__bf16*)(smem + LGBUF_OFF);
  __bf16* glds  = (__bf16*)(smem + LGBUF_OFF);     // aliases lgbuf after GEMM1
  const uint32_t bb_base = lds_off(smem + BBUF_OFF);

  const int tid = threadIdx.x;
  const int bi  = blockIdx.x;
  const int b   = bi >> 8;              // 256 blocks per batch row (T/64)
  const int t0  = (bi & 255) * 64;      // t within batch
  const long long base = (long long)bi * 64;

  // kick off async staging of the first weight tile while we fill activations
  stage_async(w1, 544, 0, bb_base, tid);

  // ---- stage x rows [t0-16, t0+64) as bf16 (single HBM read of x) ----
  for (int idx = tid; idx < 80 * 32; idx += 256) {
    int r = idx >> 5, q = idx & 31;
    int st = t0 - 16 + r;
    float4 v = make_float4(0.f, 0.f, 0.f, 0.f);
    if (st >= 0) v = *(const float4*)(x + ((long long)b * T + st) * 128 + q * 4);
    v4bf o; o[0]=(__bf16)v.x; o[1]=(__bf16)v.y; o[2]=(__bf16)v.z; o[3]=(__bf16)v.w;
    *(v4bf*)&xbuf[r * 136 + q * 4] = o;
  }
  for (int idx = tid; idx < 64 * 20; idx += 256) {
    int r = idx / 20, q = idx - r * 20;
    float4 v = *(const float4*)(lcond + (base + r) * 80 + q * 4);
    v4bf o; o[0]=(__bf16)v.x; o[1]=(__bf16)v.y; o[2]=(__bf16)v.z; o[3]=(__bf16)v.w;
    *(v4bf*)&lgbuf[r * 168 + q * 4] = o;
  }
  for (int idx = tid; idx < 64 * 16; idx += 256) {
    int r = idx >> 4, q = idx & 15;
    float4 v = *(const float4*)(gcond + (base + r) * 64 + q * 4);
    v4bf o; o[0]=(__bf16)v.x; o[1]=(__bf16)v.y; o[2]=(__bf16)v.z; o[3]=(__bf16)v.w;
    *(v4bf*)&lgbuf[r * 168 + 80 + q * 4] = o;
  }
  for (int idx = tid; idx < 64 * 6; idx += 256) {
    int r = idx / 6, q = idx - r * 6;
    v4bf z = {};
    *(v4bf*)&lgbuf[r * 168 + 144 + q * 4] = z;
  }

  const int wave = tid >> 5;
  const int lane = tid & 31;
  const int h    = lane >> 4;        // K-half select for A/B fragments
  const int lm   = lane & 15;        // M row (A) / N col (B/C)
  const int mi   = wave & 3;
  const int nj   = wave >> 2;
  const int tl   = mi * 16 + lm;

  const v8f zero = {0.f,0.f,0.f,0.f,0.f,0.f,0.f,0.f};
  v8f acc[8];
  #pragma unroll
  for (int i = 0; i < 8; ++i) acc[i] = zero;

  // ================= GEMM1: [64 x 544] x [544 x 256] =================
  #pragma unroll 1
  for (int s = 0; s < 17; ++s) {
    wait_async_barrier();                              // buffer s ready everywhere
    if (s + 1 < 17)
      stage_async(w1, 544, (s + 1) * 32, bb_base + ((s + 1) & 1) * BBUF_BYTES, tid);

    const __bf16* bcur = (const __bf16*)(smem + BBUF_OFF + (s & 1) * BBUF_BYTES);
    const int k0 = s * 32;
    const int kl = k0 + h * 8;
    const int kh = kl + 16;
    const __bf16* pa0 = (kl < 384)
        ? &xbuf[(tl + (kl >> 7) * 8) * 136 + (kl & 127)]
        : &lgbuf[tl * 168 + (kl - 384)];
    const __bf16* pa1 = (kh < 384)
        ? &xbuf[(tl + (kh >> 7) * 8) * 136 + (kh & 127)]
        : &lgbuf[tl * 168 + (kh - 384)];
    v16bf a = cat16(*(const v8bf*)pa0, *(const v8bf*)pa1);

    v16bf bf[8];
    #pragma unroll
    for (int t = 0; t < 8; ++t) {
      int nb = (t < 4) ? (nj * 64 + t * 16) : (128 + nj * 64 + (t - 4) * 16);
      const __bf16* bp = bcur + (size_t)(nb + lm) * 40 + h * 8;
      bf[t] = cat16(*(const v8bf*)bp, *(const v8bf*)(bp + 16));
    }
    #pragma unroll
    for (int t = 0; t < 8; ++t)
      acc[t] = __builtin_amdgcn_wmma_f32_16x16x32_bf16(
          false, a, false, bf[t], (short)0, acc[t], false, false);
  }

  __syncthreads();          // all waves done reading lgbuf + bbuf
  stage_async(w2, 128, 0, bb_base, tid);   // prefetch GEMM2 weights, batch 0

  // ---- gate: tanh(h) * sigmoid(s), write bf16 into glds (aliases lgbuf) ----
  #pragma unroll
  for (int p = 0; p < 4; ++p) {
    int nh = nj * 64 + p * 16 + lm;
    float bh = conv_bias[nh];
    float bs = conv_bias[nh + 128];
    #pragma unroll
    for (int r = 0; r < 8; ++r) {
      float hv = acc[p][r]     + bh;
      float sv = acc[p + 4][r] + bs;
      float gv = tanhf(hv) * (1.0f / (1.0f + __expf(-sv)));
      glds[(mi * 16 + h * 8 + r) * 136 + nh] = (__bf16)gv;
    }
  }

  // ================= GEMM2: [64 x 128] x [128 x 256] =================
  #pragma unroll
  for (int i = 0; i < 8; ++i) acc[i] = zero;
  #pragma unroll 1
  for (int s = 0; s < 4; ++s) {
    wait_async_barrier();                              // also covers glds writes at s=0
    if (s + 1 < 4)
      stage_async(w2, 128, (s + 1) * 32, bb_base + ((s + 1) & 1) * BBUF_BYTES, tid);

    const __bf16* bcur = (const __bf16*)(smem + BBUF_OFF + (s & 1) * BBUF_BYTES);
    const int k0 = s * 32;
    const __bf16* pa0 = &glds[tl * 136 + k0 + h * 8];
    v16bf a = cat16(*(const v8bf*)pa0, *(const v8bf*)(pa0 + 16));

    v16bf bf[8];
    #pragma unroll
    for (int t = 0; t < 8; ++t) {
      int nb = (t < 4) ? (nj * 64 + t * 16) : (128 + nj * 64 + (t - 4) * 16);
      const __bf16* bp = bcur + (size_t)(nb + lm) * 40 + h * 8;
      bf[t] = cat16(*(const v8bf*)bp, *(const v8bf*)(bp + 16));
    }
    #pragma unroll
    for (int t = 0; t < 8; ++t)
      acc[t] = __builtin_amdgcn_wmma_f32_16x16x32_bf16(
          false, a, false, bf[t], (short)0, acc[t], false, false);
  }

  // ---- epilogue: skip (tiles 0..3) and residual out (tiles 4..7) ----
  #pragma unroll
  for (int p = 0; p < 4; ++p) {
    int sc = nj * 64 + p * 16 + lm;
    float sb = skip_bias[sc];
    #pragma unroll
    for (int r = 0; r < 8; ++r) {
      long long row = base + mi * 16 + h * 8 + r;
      dout[TOT + row * 128 + sc] = acc[p][r] + sb;
    }
  }
  #pragma unroll
  for (int p = 4; p < 8; ++p) {
    int c = nj * 64 + (p - 4) * 16 + lm;
    float ob = out_bias[c];
    #pragma unroll
    for (int r = 0; r < 8; ++r) {
      long long row = base + mi * 16 + h * 8 + r;
      dout[row * 128 + c] = (acc[p][r] + ob + x[row * 128 + c]) * SQRT_HALF;
    }
  }
}

extern "C" void kernel_launch(void* const* d_in, const int* in_sizes, int n_in,
                              void* d_out, int out_size, void* d_ws, size_t ws_size,
                              hipStream_t stream)
{
  const float* x      = (const float*)d_in[0];
  const float* lc     = (const float*)d_in[1];
  const float* gc     = (const float*)d_in[2];
  const float* conv_k = (const float*)d_in[3];
  const float* conv_g = (const float*)d_in[4];
  const float* conv_b = (const float*)d_in[5];
  const float* lc_k   = (const float*)d_in[6];
  const float* lc_g   = (const float*)d_in[7];
  const float* gc_k   = (const float*)d_in[8];
  const float* gc_g   = (const float*)d_in[9];
  const float* skip_k = (const float*)d_in[10];
  const float* skip_g = (const float*)d_in[11];
  const float* skip_b = (const float*)d_in[12];
  const float* out_k  = (const float*)d_in[13];
  const float* out_g  = (const float*)d_in[14];
  const float* out_b  = (const float*)d_in[15];

  __bf16* w1 = (__bf16*)d_ws;                                  // 256*544 bf16
  __bf16* w2 = (__bf16*)((char*)d_ws + (size_t)256 * 544 * 2); // 256*128 bf16

  wn_prep<<<256, 128, 0, stream>>>(conv_k, conv_g, lc_k, lc_g, gc_k, gc_g,
                                   skip_k, skip_g, out_k, out_g, w1, w2);
  wavenet_main<<<2048, 256, SMEM_BYTES, stream>>>(x, lc, gc, conv_b, skip_b, out_b,
                                                  w1, w2, (float*)d_out);
}
